// ViTTransformer_66176856096999
// MI455X (gfx1250) — matrix operand order repaired and checked by reading, not compiled, not run on hardware
//
#include <hip/hip_runtime.h>

// ---------------------------------------------------------------------------
// CDNA5 WMMA (wave32) fragment types
// ---------------------------------------------------------------------------
typedef __attribute__((ext_vector_type(16))) __bf16 v16bf;
typedef __attribute__((ext_vector_type(8)))  float  v8f;
typedef __attribute__((ext_vector_type(4)))  int    v4i;

// ---------------------------------------------------------------------------
// CDNA5 async global->LDS copy path (ASYNCcnt-tracked), if the toolchain has it
// Signature (from compiler diagnostic): (v4i* as1, v4i* as3, imm offset, imm cpol)
// ---------------------------------------------------------------------------
#if defined(__has_builtin)
#if __has_builtin(__builtin_amdgcn_global_load_async_to_lds_b128)
#define USE_ASYNC_LDS 1
#endif
#endif
#ifndef USE_ASYNC_LDS
#define USE_ASYNC_LDS 0
#endif

#if USE_ASYNC_LDS
typedef __attribute__((address_space(1))) v4i gas_v4i;
typedef __attribute__((address_space(3))) v4i las_v4i;
#define COPY16(lds_ptr, g_ptr)                          \
  __builtin_amdgcn_global_load_async_to_lds_b128(       \
      (gas_v4i*)(g_ptr), (las_v4i*)(lds_ptr), 0, 0)
#if __has_builtin(__builtin_amdgcn_s_wait_asynccnt)
#define WAIT_ASYNC() __builtin_amdgcn_s_wait_asynccnt(0)
#else
#define WAIT_ASYNC() asm volatile("s_wait_asynccnt 0x0" ::: "memory")
#endif
#else
#define COPY16(lds_ptr, g_ptr) *(uint4*)(lds_ptr) = *(const uint4*)(g_ptr)
#define WAIT_ASYNC()
#endif

static __device__ __forceinline__ unsigned short f32_to_bf16(float f) {
  unsigned int u = __float_as_uint(f);
  unsigned int r = (u + 0x7FFFu + ((u >> 16) & 1u)) >> 16;  // RNE
  return (unsigned short)r;
}

static __device__ __forceinline__ v16bf pack_frag(uint4 lo, uint4 hi) {
  struct U { uint4 a, b; } u;
  u.a = lo; u.b = hi;
  return __builtin_bit_cast(v16bf, u);
}

// ---------------------------------------------------------------------------
// Transposed fp32 -> bf16 weight conversion: in[R][C] f32 -> out[C][R] bf16
// (gives the GEMM a K-contiguous B operand so B fragments are b128 loads)
// ---------------------------------------------------------------------------
__global__ __launch_bounds__(256) void transpose_convert_bf16(
    const float* __restrict__ in, unsigned short* __restrict__ out,
    int R, int C) {
  __shared__ float tile[32][33];
  int tx = threadIdx.x & 31, ty = threadIdx.x >> 5;  // 32 x 8
  int r0 = blockIdx.y * 32, c0 = blockIdx.x * 32;
#pragma unroll
  for (int j = 0; j < 4; ++j)
    tile[ty + j * 8][tx] = in[(size_t)(r0 + ty + j * 8) * C + c0 + tx];
  __syncthreads();
#pragma unroll
  for (int j = 0; j < 4; ++j)
    out[(size_t)(c0 + ty + j * 8) * R + r0 + tx] =
        f32_to_bf16(tile[tx][ty + j * 8]);
}

// ---------------------------------------------------------------------------
// LayerNorm over D=768, one row per block (256 threads = 8 waves), bf16 out
// ---------------------------------------------------------------------------
__global__ __launch_bounds__(256) void layernorm_bf16(
    const float* __restrict__ x, const float* __restrict__ g,
    const float* __restrict__ b, unsigned short* __restrict__ out) {
  const int D = 768;
  int row = blockIdx.x, tid = threadIdx.x;
  int lane = tid & 31, wid = tid >> 5;
  const float* xr = x + (size_t)row * D;

  float v0 = xr[tid], v1 = xr[tid + 256], v2 = xr[tid + 512];
  float s  = v0 + v1 + v2;
  float s2 = v0 * v0 + v1 * v1 + v2 * v2;
#pragma unroll
  for (int o = 16; o > 0; o >>= 1) {
    s  += __shfl_xor(s,  o, 32);
    s2 += __shfl_xor(s2, o, 32);
  }
  __shared__ float ps[8], ps2[8];
  __shared__ float mu_s, rs_s;
  if (lane == 0) { ps[wid] = s; ps2[wid] = s2; }
  __syncthreads();
  if (tid == 0) {
    float a = 0.f, a2 = 0.f;
#pragma unroll
    for (int i = 0; i < 8; ++i) { a += ps[i]; a2 += ps2[i]; }
    float mu  = a * (1.0f / 768.0f);
    float var = a2 * (1.0f / 768.0f) - mu * mu;
    mu_s = mu;
    rs_s = rsqrtf(var + 1e-5f);
  }
  __syncthreads();
  float mu = mu_s, rs = rs_s;
  size_t ob = (size_t)row * D;
  out[ob + tid]       = f32_to_bf16((v0 - mu) * rs * g[tid]       + b[tid]);
  out[ob + tid + 256] = f32_to_bf16((v1 - mu) * rs * g[tid + 256] + b[tid + 256]);
  out[ob + tid + 512] = f32_to_bf16((v2 - mu) * rs * g[tid + 512] + b[tid + 512]);
}

// ---------------------------------------------------------------------------
// WMMA bf16 GEMM: C = A[MxK] * Bt[NxK]^T (+bias) (+residual) (optional relu)
// A row-major bf16; Bt is the transposed weight [N][K] bf16.
// Block tile 128x128, BK=32, double-buffered LDS (async prefetch if available)
// 8 waves, each wave computes 32x64 = 2x4 WMMA 16x16 tiles.
// M%128==0, N%128==0, K%32==0 (holds for all shapes here).
// ---------------------------------------------------------------------------
__global__ __launch_bounds__(256) void gemm_bf16_wmma(
    const unsigned short* __restrict__ A, const unsigned short* __restrict__ Bt,
    const float* __restrict__ bias, const float* __restrict__ residual,
    float* __restrict__ outF, unsigned short* __restrict__ outB,
    int M, int N, int K, int relu) {
  const int BM = 128, BN = 128, BK = 32;
  __shared__ __align__(16) unsigned short As[2][BM * BK];  // 2 x 8 KB
  __shared__ __align__(16) unsigned short Bs[2][BN * BK];  // 2 x 8 KB

  int tid  = threadIdx.x;
  int lane = tid & 31, wid = tid >> 5;
  int row0 = blockIdx.y * BM;
  int col0 = blockIdx.x * BN;
  int wm = (wid & 3) * 32;    // wave M offset inside block tile
  int wn = (wid >> 2) * 64;   // wave N offset inside block tile

  // Fragment lane decomposition (ISA 7.12.2, 16-bit A 16x32 / B 32x16)
  int rA  = lane & 15;            // A fragment row
  int kgA = (lane >> 4) << 3;     // 0 or 8   (A K sub-group)
  int cB  = lane & 15;            // B fragment column
  int kgB = (lane >> 4) << 3;     // 0 or 8   (B K sub-group)

  // Per-thread staging chunks: both tiles are 128 rows x 32 k, 8-elem chunks
  int r0c = tid >> 2,          c0c = (tid & 3) * 8;
  int r1c = (tid + 256) >> 2,  c1c = ((tid + 256) & 3) * 8;

  v8f acc[2][4];
#pragma unroll
  for (int mi = 0; mi < 2; ++mi)
#pragma unroll
    for (int ni = 0; ni < 4; ++ni) acc[mi][ni] = (v8f)(0.0f);

  auto stage = [&](int buf, int k0) {
    COPY16(&As[buf][r0c * BK + c0c], &A [(size_t)(row0 + r0c) * K + k0 + c0c]);
    COPY16(&As[buf][r1c * BK + c1c], &A [(size_t)(row0 + r1c) * K + k0 + c1c]);
    COPY16(&Bs[buf][r0c * BK + c0c], &Bt[(size_t)(col0 + r0c) * K + k0 + c0c]);
    COPY16(&Bs[buf][r1c * BK + c1c], &Bt[(size_t)(col0 + r1c) * K + k0 + c1c]);
  };

  stage(0, 0);
  WAIT_ASYNC();
  __syncthreads();

  int nk = K / BK;
  int buf = 0;
  for (int kt = 0; kt < nk; ++kt) {
    if (kt + 1 < nk) stage(buf ^ 1, (kt + 1) * BK);

    v16bf afrag[2];
#pragma unroll
    for (int mi = 0; mi < 2; ++mi) {
      int base = (wm + mi * 16 + rA) * BK;
      uint4 lo = *(const uint4*)&As[buf][base + kgA];
      uint4 hi = *(const uint4*)&As[buf][base + 16 + kgA];
      afrag[mi] = pack_frag(lo, hi);
    }
    v16bf bfrag[4];
#pragma unroll
    for (int ni = 0; ni < 4; ++ni) {
      int bbase = (wn + ni * 16 + cB) * BK + kgB;  // K kgB..kgB+7, 16+kgB..16+kgB+7
      uint4 lo = *(const uint4*)&Bs[buf][bbase];
      uint4 hi = *(const uint4*)&Bs[buf][bbase + 16];
      bfrag[ni] = pack_frag(lo, hi);
    }

#pragma unroll
    for (int mi = 0; mi < 2; ++mi)
#pragma unroll
      for (int ni = 0; ni < 4; ++ni)
        acc[mi][ni] = __builtin_amdgcn_wmma_f32_16x16x32_bf16(
            false, afrag[mi], false, bfrag[ni], (short)0, acc[mi][ni],
            false, false);

    WAIT_ASYNC();
    __syncthreads();
    buf ^= 1;
  }

  // Epilogue: C/D layout — element i => row i (lanes 0-15) / row 8+i (16-31)
  int lr = (lane >> 4) << 3;
  int lc = lane & 15;
#pragma unroll
  for (int mi = 0; mi < 2; ++mi) {
#pragma unroll
    for (int ni = 0; ni < 4; ++ni) {
      int gcol = col0 + wn + ni * 16 + lc;
      float bv = bias ? bias[gcol] : 0.0f;
#pragma unroll
      for (int i = 0; i < 8; ++i) {
        int grow = row0 + wm + mi * 16 + lr + i;
        float v = acc[mi][ni][i] + bv;
        if (residual) v += residual[(size_t)grow * N + gcol];
        if (relu) v = fmaxf(v, 0.0f);
        if (outF) outF[(size_t)grow * N + gcol] = v;
        else      outB[(size_t)grow * N + gcol] = f32_to_bf16(v);
      }
    }
  }
}

// ---------------------------------------------------------------------------
// Attention: 96 heads of dim 8, N=512. One workgroup per (head, batch).
// K,V head slices + per-wave score rows in LDS; two-pass softmax per row,
// wave32 shfl reductions. ctx written as bf16 for the out-proj GEMM.
// ---------------------------------------------------------------------------
__global__ __launch_bounds__(256) void attention_head(
    const float* __restrict__ q, const float* __restrict__ k,
    const float* __restrict__ v, unsigned short* __restrict__ ctx) {
  const int N = 512, D = 768, Dh = 8;
  int head = blockIdx.x, batch = blockIdx.y;
  int tid = threadIdx.x, lane = tid & 31, wid = tid >> 5;

  __shared__ float Ksh[512 * 8];   // 16 KB
  __shared__ float Vsh[512 * 8];   // 16 KB
  __shared__ float Ssh[8 * 512];   // 16 KB (per-wave score rows)

  size_t base = (size_t)batch * N * D + (size_t)head * Dh;
  for (int c = tid; c < 1024; c += 256) {
    int row = c >> 1, part = (c & 1) * 4;
    *(float4*)&Ksh[row * 8 + part] =
        *(const float4*)&k[base + (size_t)row * D + part];
    *(float4*)&Vsh[row * 8 + part] =
        *(const float4*)&v[base + (size_t)row * D + part];
  }
  __syncthreads();

  const float invs = 0.10206207261596575f;  // 1/sqrt(96)
  for (int row = wid; row < N; row += 8) {
    const float* qr = &q[base + (size_t)row * D];
    float qv[8];
#pragma unroll
    for (int d = 0; d < 8; ++d) qv[d] = qr[d];

    float m = -3.0e38f;
    for (int c = lane; c < N; c += 32) {
      const float* kp = &Ksh[c * 8];
      float s = 0.0f;
#pragma unroll
      for (int d = 0; d < 8; ++d) s += qv[d] * kp[d];
      s *= invs;
      Ssh[wid * 512 + c] = s;
      m = fmaxf(m, s);
    }
#pragma unroll
    for (int o = 16; o > 0; o >>= 1) m = fmaxf(m, __shfl_xor(m, o, 32));

    float sum = 0.0f;
    float a[8];
#pragma unroll
    for (int d = 0; d < 8; ++d) a[d] = 0.0f;
    for (int c = lane; c < N; c += 32) {
      float e = __expf(Ssh[wid * 512 + c] - m);
      sum += e;
      const float* vp = &Vsh[c * 8];
#pragma unroll
      for (int d = 0; d < 8; ++d) a[d] += e * vp[d];
    }
#pragma unroll
    for (int o = 16; o > 0; o >>= 1) {
      sum += __shfl_xor(sum, o, 32);
#pragma unroll
      for (int d = 0; d < 8; ++d) a[d] += __shfl_xor(a[d], o, 32);
    }
    float inv = 1.0f / sum;
    size_t ob = base + (size_t)row * D;
#pragma unroll
    for (int d = 0; d < 8; ++d)
      if (lane == d) ctx[ob + d] = f32_to_bf16(a[d] * inv);
  }
}

// ---------------------------------------------------------------------------
// Host orchestration (graph-capture safe: only kernel launches on `stream`)
// ---------------------------------------------------------------------------
extern "C" void kernel_launch(void* const* d_in, const int* in_sizes, int n_in,
                              void* d_out, int out_size, void* d_ws, size_t ws_size,
                              hipStream_t stream) {
  (void)in_sizes; (void)n_in; (void)out_size; (void)ws_size;
  const int B = 4, N = 512, D = 768, F = 2048;
  const int M = B * N;  // 2048 rows

  const float* x_in = (const float*)d_in[0];
  const float* wq = (const float*)d_in[1];  const float* bq = (const float*)d_in[2];
  const float* wk = (const float*)d_in[3];  const float* bk = (const float*)d_in[4];
  const float* wv = (const float*)d_in[5];  const float* bv = (const float*)d_in[6];
  const float* wo = (const float*)d_in[7];  const float* bo = (const float*)d_in[8];
  const float* ln1g = (const float*)d_in[9];  const float* ln1b = (const float*)d_in[10];
  const float* ln2g = (const float*)d_in[11]; const float* ln2b = (const float*)d_in[12];
  const float* w1 = (const float*)d_in[13]; const float* b1 = (const float*)d_in[14];
  const float* w2 = (const float*)d_in[15]; const float* b2 = (const float*)d_in[16];

  char* ws = (char*)d_ws;
  size_t off = 0;
  auto alloc = [&](size_t bytes) -> void* {
    void* p = ws + off;
    off += (bytes + 255) & ~(size_t)255;
    return p;
  };

  // Transposed bf16 weights: [N][K] layout for the GEMM B operand
  unsigned short* wq_t  = (unsigned short*)alloc((size_t)D * D * 2);
  unsigned short* wk_t  = (unsigned short*)alloc((size_t)D * D * 2);
  unsigned short* wv_t  = (unsigned short*)alloc((size_t)D * D * 2);
  unsigned short* wo_t  = (unsigned short*)alloc((size_t)D * D * 2);
  unsigned short* w1_t  = (unsigned short*)alloc((size_t)D * F * 2);  // [F][D]
  unsigned short* w2_t  = (unsigned short*)alloc((size_t)F * D * 2);  // [D][F]
  unsigned short* h_b   = (unsigned short*)alloc((size_t)M * D * 2);
  float*          qf    = (float*)alloc((size_t)M * D * 4);
  float*          kf    = (float*)alloc((size_t)M * D * 4);
  float*          vf    = (float*)alloc((size_t)M * D * 4);
  unsigned short* ctx_b = (unsigned short*)alloc((size_t)M * D * 2);
  float*          x1    = (float*)alloc((size_t)M * D * 4);
  unsigned short* t_b   = (unsigned short*)alloc((size_t)M * F * 2);
  float*          xbuf  = (float*)alloc((size_t)M * D * 4);

  // Weight transpose+convert fp32 -> bf16 (deterministic, every call)
  {
    dim3 gDD(D / 32, D / 32);      // in [D][D]
    dim3 gDF(F / 32, D / 32);      // in [D][F] -> out [F][D]
    dim3 gFD(D / 32, F / 32);      // in [F][D] -> out [D][F]
    transpose_convert_bf16<<<gDD, 256, 0, stream>>>(wq, wq_t, D, D);
    transpose_convert_bf16<<<gDD, 256, 0, stream>>>(wk, wk_t, D, D);
    transpose_convert_bf16<<<gDD, 256, 0, stream>>>(wv, wv_t, D, D);
    transpose_convert_bf16<<<gDD, 256, 0, stream>>>(wo, wo_t, D, D);
    transpose_convert_bf16<<<gDF, 256, 0, stream>>>(w1, w1_t, D, F);
    transpose_convert_bf16<<<gFD, 256, 0, stream>>>(w2, w2_t, F, D);
  }

  dim3 gD(D / 128, M / 128);   // (6, 16)  N=768 outputs
  dim3 gF(F / 128, M / 128);   // (16, 16) N=2048 outputs
  dim3 gAttn(96, 4);

  const float* xcur = x_in;
  for (int it = 0; it < 4; ++it) {
    // LN1 -> h (bf16)
    layernorm_bf16<<<M, 256, 0, stream>>>(xcur, ln1g, ln1b, h_b);
    // Q, K, V projections (fp32 out)
    gemm_bf16_wmma<<<gD, 256, 0, stream>>>(h_b, wq_t, bq, nullptr, qf, nullptr, M, D, D, 0);
    gemm_bf16_wmma<<<gD, 256, 0, stream>>>(h_b, wk_t, bk, nullptr, kf, nullptr, M, D, D, 0);
    gemm_bf16_wmma<<<gD, 256, 0, stream>>>(h_b, wv_t, bv, nullptr, vf, nullptr, M, D, D, 0);
    // Attention -> ctx (bf16)
    attention_head<<<gAttn, 256, 0, stream>>>(qf, kf, vf, ctx_b);
    // Out-proj + residual(xcur) -> x1 (fp32)
    gemm_bf16_wmma<<<gD, 256, 0, stream>>>(ctx_b, wo_t, bo, xcur, x1, nullptr, M, D, D, 0);
    // LN2 -> h (bf16)
    layernorm_bf16<<<M, 256, 0, stream>>>(x1, ln2g, ln2b, h_b);
    // FFN up + ReLU -> t (bf16)
    gemm_bf16_wmma<<<gF, 256, 0, stream>>>(h_b, w1_t, b1, nullptr, nullptr, t_b, M, F, D, 1);
    // FFN down + residual(x1) -> next x (fp32); last iter writes d_out
    float* xnext = (it == 3) ? (float*)d_out : xbuf;
    gemm_bf16_wmma<<<gD, 256, 0, stream>>>(t_b, w2_t, b2, x1, xnext, nullptr, M, D, F, 0);
    xcur = xnext;
  }
}
